// SelfAttention_58531814310020
// MI455X (gfx1250) — matrix operand order repaired and checked by reading, not compiled, not run on hardware
//
#include <hip/hip_runtime.h>
#include <hip/hip_fp16.h>

// ---------------------------------------------------------------------------
// Self-attention for B=8, C=256, H=W=64 (N=4096, d=32) on gfx1250 (MI455X).
// Phase 0: x -> xT[N][C] f16 (LDS-tiled transpose), W* -> f16.
// Phase 1: ALL projections on WMMA:
//   q,k: D[kd][n] = W(16x32-chunks) x xT   -> packed b128 stores to qT/kT[N][32]
//   v:   D[n][c]  = xT x Wv(K x N view)    -> packed b128 stores to v[C][N]
// Phase 2: flash-attention, 4 waves/block share one batch; V tiles double-
//   buffered into LDS via global_load_async_to_lds_b128 (ASYNCcnt); softmax
//   row reductions are single-instruction DPP16 VOP2 ops.
// ---------------------------------------------------------------------------

constexpr int kB = 8;
constexpr int kC = 256;
constexpr int kD = 32;     // head dim == WMMA K for f16  (16x16x32)
constexpr int kN = 4096;   // H*W
constexpr int kVP = 40;    // LDS V row pitch in halfs (32 keys + 8 pad)

typedef __attribute__((ext_vector_type(16))) _Float16 v16h;
typedef __attribute__((ext_vector_type(8)))  float    v8f;

union V16H {
  v16h  v;
  uint4 u[2];
};
union PK8 {
  uint4    u;
  _Float16 h[8];
};

#if defined(__gfx1250__) &&                                                   \
    __has_builtin(__builtin_amdgcn_global_load_async_to_lds_b128) &&          \
    __has_builtin(__builtin_amdgcn_s_wait_asynccnt)
#define USE_ASYNC_V 1
#else
#define USE_ASYNC_V 0
#endif

// --- DPP16 16-lane reductions: one VOP2+DPP instruction per step -----------
__device__ __forceinline__ float rowmax16(float v) {
  float r;
  asm("v_max_num_f32_dpp %0, %1, %1 quad_perm:[1,0,3,2] row_mask:0xf bank_mask:0xf bound_ctrl:1"
      : "=v"(r) : "v"(v));
  asm("v_max_num_f32_dpp %0, %1, %1 quad_perm:[2,3,0,1] row_mask:0xf bank_mask:0xf bound_ctrl:1"
      : "=v"(v) : "v"(r));
  asm("v_max_num_f32_dpp %0, %1, %1 row_half_mirror row_mask:0xf bank_mask:0xf bound_ctrl:1"
      : "=v"(r) : "v"(v));
  asm("v_max_num_f32_dpp %0, %1, %1 row_mirror row_mask:0xf bank_mask:0xf bound_ctrl:1"
      : "=v"(v) : "v"(r));
  return v;
}
__device__ __forceinline__ float rowsum16(float v) {
  float r;
  asm("v_add_f32_dpp %0, %1, %1 quad_perm:[1,0,3,2] row_mask:0xf bank_mask:0xf bound_ctrl:1"
      : "=v"(r) : "v"(v));
  asm("v_add_f32_dpp %0, %1, %1 quad_perm:[2,3,0,1] row_mask:0xf bank_mask:0xf bound_ctrl:1"
      : "=v"(v) : "v"(r));
  asm("v_add_f32_dpp %0, %1, %1 row_half_mirror row_mask:0xf bank_mask:0xf bound_ctrl:1"
      : "=v"(r) : "v"(v));
  asm("v_add_f32_dpp %0, %1, %1 row_mirror row_mask:0xf bank_mask:0xf bound_ctrl:1"
      : "=v"(v) : "v"(r));
  return v;
}

// ---------------------------------------------------------------------------
// Phase 0a: f32 -> f16 weight conversion (Wq/Wk: 32x256, Wv: 256x256).
// ---------------------------------------------------------------------------
__global__ __launch_bounds__(256) void wcvt_kernel(
    const float* __restrict__ Wq, const float* __restrict__ Wk,
    const float* __restrict__ Wv, _Float16* __restrict__ Wqh,
    _Float16* __restrict__ Wkh, _Float16* __restrict__ Wvh) {
  int idx = blockIdx.x * 256 + threadIdx.x;
  if (idx < kD * kC) {
    Wqh[idx] = (_Float16)Wq[idx];
    Wkh[idx] = (_Float16)Wk[idx];
  }
  if (idx < kC * kC) Wvh[idx] = (_Float16)Wv[idx];
}

// ---------------------------------------------------------------------------
// Phase 0b: xT[b][n][c] f16 = transpose(x[b][c][n]).  64x64 LDS tiles,
// coalesced on both the f32 read and the packed-f16 write.
// ---------------------------------------------------------------------------
__global__ __launch_bounds__(256) void xT_kernel(
    const float* __restrict__ x, _Float16* __restrict__ xT) {
  __shared__ __attribute__((aligned(16))) float tile[64][65];
  const int tid  = threadIdx.x;
  const int nblk = kN / 64;  // 64
  const int cblk = kC / 64;  // 4
  const int b  = blockIdx.x / (nblk * cblk);
  const int rm = blockIdx.x % (nblk * cblk);
  const int c0 = (rm / nblk) * 64;
  const int n0 = (rm % nblk) * 64;

  const float* xb = x + ((size_t)b * kC + c0) * kN + n0;
#pragma unroll
  for (int k = 0; k < 16; ++k) {
    int cr = (tid >> 6) + k * 4;
    tile[cr][tid & 63] = xb[(size_t)cr * kN + (tid & 63)];
  }
  __syncthreads();

  const int nr = tid >> 2;  // 0..63  token within tile
  const int ck = tid & 3;   // 0..3   16-channel segment
  union { uint4 u[2]; _Float16 h[16]; } pk;
#pragma unroll
  for (int i = 0; i < 16; ++i) pk.h[i] = (_Float16)tile[ck * 16 + i][nr];
  _Float16* dst = xT + ((size_t)b * kN + n0 + nr) * kC + c0 + ck * 16;
  *(uint4*)(dst)     = pk.u[0];
  *(uint4*)(dst + 8) = pk.u[1];
}

// ---------------------------------------------------------------------------
// Phase 1: WMMA projections.  One wave per (b, 16-token tile); xT row chunks
// cached in registers in BOTH operand stripings; 160 WMMAs per wave.
// ---------------------------------------------------------------------------
__global__ __launch_bounds__(128, 1) void proj_gemm_kernel(
    const _Float16* __restrict__ xT, const _Float16* __restrict__ Wqh,
    const _Float16* __restrict__ Wkh, const _Float16* __restrict__ Wvh,
    const float* __restrict__ bq, const float* __restrict__ bk,
    const float* __restrict__ bv, _Float16* __restrict__ qT,
    _Float16* __restrict__ kT, _Float16* __restrict__ vh) {
  const int lane = threadIdx.x & 31;
  const int wave = threadIdx.x >> 5;
  const int wid  = blockIdx.x * 4 + wave;  // 2048 waves
  const int b    = wid >> 8;
  const int n0   = (wid & 255) << 4;
  const int lh = lane & 15, hh = lane >> 4;

  const _Float16* xrow = xT + ((size_t)b * kN + n0 + lh) * kC;

  // cache this token-tile's xT rows in both WMMA stripings
  V16H Bx[8], Ax[8];
#pragma unroll
  for (int kk = 0; kk < 8; ++kk) {
    const _Float16* pB = xrow + kk * 32 + hh * 16;
    Bx[kk].u[0] = *(const uint4*)(pB);
    Bx[kk].u[1] = *(const uint4*)(pB + 8);
    const _Float16* pA = xrow + kk * 32 + hh * 16;
    Ax[kk].u[0] = *(const uint4*)(pA);
    Ax[kk].u[1] = *(const uint4*)(pA + 8);
  }
  const v8f vzero = {};

  // --- q,k: D[kd][n] = W x xT; lane packs 8 consecutive kd -> 1 b128 store --
#pragma unroll 1
  for (int wsel = 0; wsel < 4; ++wsel) {
    const _Float16* W  = (wsel & 2) ? Wkh : Wqh;
    const float*    bb = (wsel & 2) ? bk : bq;
    _Float16*       dT = (wsel & 2) ? kT : qT;
    const int kd0 = (wsel & 1) << 4;
    v8f acc = vzero;
#pragma unroll
    for (int kk = 0; kk < 8; ++kk) {
      V16H Aw;
      const _Float16* pA = W + (size_t)(kd0 + lh) * kC + kk * 32 + hh * 16;
      Aw.u[0] = *(const uint4*)(pA);
      Aw.u[1] = *(const uint4*)(pA + 8);
      acc = __builtin_amdgcn_wmma_f32_16x16x32_f16(
          false, Aw.v, false, Bx[kk].v, (short)0, acc, false, false);
    }
    const float* bp = bb + kd0 + 8 * hh;   // 8 consecutive bias values
    PK8 pk;
#pragma unroll
    for (int r = 0; r < 8; ++r) pk.h[r] = (_Float16)(acc[r] + bp[r]);
    *(uint4*)(dT + ((size_t)b * kN + n0 + lh) * kD + kd0 + 8 * hh) = pk.u;
  }

  // --- v: D[n][c] = xT x Wv(KxN view); lane packs 8 consecutive n ----------
#pragma unroll 1
  for (int ct = 0; ct < 16; ++ct) {
    const int c0  = ct << 4;
    const float bvv = bv[c0 + lh];
    v8f acc = vzero;
#pragma unroll
    for (int kk = 0; kk < 8; ++kk) {
      V16H Bw;
      const _Float16* pB = Wvh + (size_t)(c0 + lh) * kC + kk * 32 + hh * 16;
      Bw.u[0] = *(const uint4*)(pB);
      Bw.u[1] = *(const uint4*)(pB + 8);
      acc = __builtin_amdgcn_wmma_f32_16x16x32_f16(
          false, Ax[kk].v, false, Bw.v, (short)0, acc, false, false);
    }
    PK8 pk;
#pragma unroll
    for (int r = 0; r < 8; ++r) pk.h[r] = (_Float16)(acc[r] + bvv);
    *(uint4*)(vh + ((size_t)b * kC + c0 + lh) * kN + n0 + 8 * hh) = pk.u;
  }
}

// ---------------------------------------------------------------------------
// Phase 2: WMMA flash attention (unchanged from previous round).
// ---------------------------------------------------------------------------
#if USE_ASYNC_V
typedef int v4i_vec __attribute__((vector_size(16)));
typedef __attribute__((address_space(1))) v4i_vec glob_v4i;
typedef __attribute__((address_space(3))) v4i_vec lds_v4i;

__device__ __forceinline__ void async_load_v_tile(
    const _Float16* __restrict__ vbase, int j0, _Float16* vbuf, int tid) {
  // 256 channels x 32 keys x 2B = 16 KB = 1024 b128 chunks; 8 per thread.
#pragma unroll
  for (int q = 0; q < 8; ++q) {
    int id = q * 128 + tid;
    int c  = id >> 2;
    int ck = id & 3;
    const _Float16* g = vbase + (size_t)c * kN + j0 + ck * 8;
    _Float16*       l = vbuf + c * kVP + ck * 8;
    __builtin_amdgcn_global_load_async_to_lds_b128(
        (glob_v4i*)g, (lds_v4i*)l, 0, 0);
  }
}
#endif

__global__ __launch_bounds__(128, 1) void attn_kernel(
    const _Float16* __restrict__ qT, const _Float16* __restrict__ kT,
    const _Float16* __restrict__ vh, float* __restrict__ out) {
  __shared__ __attribute__((aligned(16))) _Float16 plds_all[4 * 16 * 32];
#if USE_ASYNC_V
  __shared__ __attribute__((aligned(16))) _Float16 vlds[2][kC * kVP];  // 40 KB
#endif

  const int tid  = threadIdx.x;
  const int lane = tid & 31;
  const int wave = tid >> 5;
  _Float16* plds = plds_all + wave * (16 * 32);

  const int b  = blockIdx.x >> 6;                    // 64 blocks per batch
  const int i0 = ((blockIdx.x & 63) * 4 + wave) << 4;

  const int lh = lane & 15;
  const int hh = lane >> 4;

  V16H Qa;
  {
    const _Float16* qrow = qT + ((size_t)b * kN + i0 + lh) * kD + hh * 16;
    Qa.u[0] = *(const uint4*)(qrow);
    Qa.u[1] = *(const uint4*)(qrow + 8);
  }

  v8f acc[16];
  const v8f vzero = {};
#pragma unroll
  for (int ct = 0; ct < 16; ++ct) acc[ct] = vzero;
  float m[8], l[8];
#pragma unroll
  for (int r = 0; r < 8; ++r) { m[r] = -3.0e38f; l[r] = 0.f; }

  const _Float16* kbase = kT + (size_t)b * kN * kD;
  const _Float16* vbase = vh + (size_t)b * kC * kN;

#if USE_ASYNC_V
  async_load_v_tile(vbase, 0, &vlds[0][0], tid);  // prime first buffer
#endif

  for (int t = 0; t < kN / 32; ++t) {
    const int j0  = t * 32;
    const int cur = t & 1;
    (void)cur;

    const _Float16* kr0 = kbase + (size_t)(j0 + lh) * kD + hh * 16;
    V16H Kb0, Kb1;
    Kb0.u[0] = *(const uint4*)(kr0);
    Kb0.u[1] = *(const uint4*)(kr0 + 8);
    const _Float16* kr1 = kr0 + 16 * kD;
    Kb1.u[0] = *(const uint4*)(kr1);
    Kb1.u[1] = *(const uint4*)(kr1 + 8);
    __builtin_prefetch(kr0 + 32 * kD, 0, 3);  // next iteration's K rows

    v8f S0 = __builtin_amdgcn_wmma_f32_16x16x32_f16(
        false, Qa.v, false, Kb0.v, (short)0, vzero, false, false);
    v8f S1 = __builtin_amdgcn_wmma_f32_16x16x32_f16(
        false, Qa.v, false, Kb1.v, (short)0, vzero, false, false);

    float scale[8];
    bool  rescale = false;
#pragma unroll
    for (int r = 0; r < 8; ++r) {
      float s0 = S0[r], s1 = S1[r];
      float mloc = rowmax16(fmaxf(s0, s1));
      float mnew = fmaxf(m[r], mloc);
      float sc   = __expf(m[r] - mnew);
      float p0   = __expf(s0 - mnew);
      float p1   = __expf(s1 - mnew);
      float ps   = rowsum16(p0 + p1);
      l[r]     = l[r] * sc + ps;
      m[r]     = mnew;
      scale[r] = sc;
      rescale |= (sc != 1.0f);
      int row = r + (hh << 3);
      plds[row * 32 + lh]      = (_Float16)p0;
      plds[row * 32 + 16 + lh] = (_Float16)p1;
    }

    if (__ballot(rescale) != 0ull) {
#pragma unroll
      for (int ct = 0; ct < 16; ++ct)
#pragma unroll
        for (int r = 0; r < 8; ++r) acc[ct][r] *= scale[r];
    }

#if USE_ASYNC_V
    __builtin_amdgcn_s_wait_asynccnt(0);
    __syncthreads();
    if (t + 1 < kN / 32)
      async_load_v_tile(vbase, j0 + 32, &vlds[cur ^ 1][0], tid);
#endif

    V16H Pa;
    {
      const _Float16* pr = plds + lh * 32 + hh * 16;
      Pa.u[0] = *(const uint4*)(pr);
      Pa.u[1] = *(const uint4*)(pr + 8);
    }

#pragma unroll
    for (int ct = 0; ct < 16; ++ct) {
      V16H Vb;
#if USE_ASYNC_V
      const _Float16* vr = &vlds[cur][0] + (ct * 16 + lh) * kVP + hh * 16;
#else
      const _Float16* vr = vbase + (size_t)(ct * 16 + lh) * kN + j0 + hh * 16;
#endif
      Vb.u[0] = *(const uint4*)(vr);
      Vb.u[1] = *(const uint4*)(vr + 8);
      acc[ct] = __builtin_amdgcn_wmma_f32_16x16x32_f16(
          false, Pa.v, false, Vb.v, (short)0, acc[ct], false, false);
    }
  }

#pragma unroll
  for (int r = 0; r < 8; ++r) {
    float inv = 1.f / l[r];
    int i = i0 + r + (hh << 3);
#pragma unroll
    for (int ct = 0; ct < 16; ++ct) {
      int c = ct * 16 + lh;
      out[((size_t)b * kC + c) * kN + i] = acc[ct][r] * inv;
    }
  }
}

// ---------------------------------------------------------------------------
extern "C" void kernel_launch(void* const* d_in, const int* in_sizes, int n_in,
                              void* d_out, int out_size, void* d_ws,
                              size_t ws_size, hipStream_t stream) {
  (void)in_sizes; (void)n_in; (void)out_size; (void)ws_size;
  const float* x  = (const float*)d_in[0];
  const float* Wq = (const float*)d_in[1];
  const float* bq = (const float*)d_in[2];
  const float* Wk = (const float*)d_in[3];
  const float* bk = (const float*)d_in[4];
  const float* Wv = (const float*)d_in[5];
  const float* bv = (const float*)d_in[6];
  float* out = (float*)d_out;

  // workspace layout (halfs)
  const size_t qkSz = (size_t)kB * kN * kD;   // 1 Mi halfs
  const size_t vSz  = (size_t)kB * kC * kN;   // 8 Mi halfs
  _Float16* qT  = (_Float16*)d_ws;
  _Float16* kT  = qT + qkSz;
  _Float16* vh  = kT + qkSz;
  _Float16* xTh = vh + vSz;                   // 8 Mi halfs
  _Float16* Wqh = xTh + vSz;
  _Float16* Wkh = Wqh + (size_t)kD * kC;
  _Float16* Wvh = Wkh + (size_t)kD * kC;      // + 64 Ki halfs

  wcvt_kernel<<<(kC * kC) / 256, 256, 0, stream>>>(Wq, Wk, Wv, Wqh, Wkh, Wvh);
  xT_kernel<<<kB * (kC / 64) * (kN / 64), 256, 0, stream>>>(x, xTh);
  proj_gemm_kernel<<<(kB * (kN / 16)) / 4, 128, 0, stream>>>(
      xTh, Wqh, Wkh, Wvh, bq, bk, bv, qT, kT, vh);
  attn_kernel<<<(kB * (kN / 16)) / 4, 128, 0, stream>>>(qT, kT, vh, out);
}
